// ArithmeticModel_32727650795903
// MI455X (gfx1250) — matrix operand order repaired and checked by reading, not compiled, not run on hardware
//
#include <hip/hip_runtime.h>

// ---------------------------------------------------------------------------
// Types for CDNA5 WMMA (wave32, 16x16x32 f16 -> f32)
// ---------------------------------------------------------------------------
typedef __attribute__((ext_vector_type(16))) _Float16 v16h;
typedef __attribute__((ext_vector_type(8)))  _Float16 v8h;
typedef __attribute__((ext_vector_type(4)))  _Float16 v4h;
typedef __attribute__((ext_vector_type(8)))  float    v8f;

#define BB 2
#define SS 1024
#define VV 32000
#define DD 1024
#define LL 8
#define HH 16
#define FF 4096
#define DH 64
#define TT (BB * SS)   // 2048 tokens

#define WMMA_F16(A, B, C) \
  __builtin_amdgcn_wmma_f32_16x16x32_f16(false, (A), false, (B), (short)0, (C), false, false)

// Fuse two contiguous 8-half (b128) LDS/global loads into one v16h fragment.
__device__ __forceinline__ v16h ld_frag2(const _Float16* p0, const _Float16* p1) {
  v8h lo = *(const v8h*)p0;
  v8h hi = *(const v8h*)p1;
  return __builtin_shufflevector(lo, hi, 0, 1, 2, 3, 4, 5, 6, 7,
                                 8, 9, 10, 11, 12, 13, 14, 15);
}

// ---------------------------------------------------------------------------
// f32 -> f16 converter (n must be a multiple of 4)
// ---------------------------------------------------------------------------
__global__ __launch_bounds__(256) void cvt_f32_f16(const float* __restrict__ in,
                                                   _Float16* __restrict__ out, int n) {
  int i = (blockIdx.x * 256 + threadIdx.x) * 4;
  if (i < n) {
    float4 v = *(const float4*)(in + i);
    v4h h = {(_Float16)v.x, (_Float16)v.y, (_Float16)v.z, (_Float16)v.w};
    *(v4h*)(out + i) = h;
  }
}

// ---------------------------------------------------------------------------
// Transpose + convert: Wt[n][k] (f16, K-major) = W[k][n] (f32, N-major).
// 32x32 tiles through LDS; blockIdx.z selects the layer (stride K*N).
// ---------------------------------------------------------------------------
__global__ __launch_bounds__(256) void transpose_cvt(const float* __restrict__ W,
                                                     _Float16* __restrict__ Wt,
                                                     int K, int N) {
  __shared__ float tile[32][33];
  const float* Wl  = W  + (size_t)blockIdx.z * K * N;
  _Float16*   Wtl  = Wt + (size_t)blockIdx.z * K * N;
  int k0 = blockIdx.x * 32, n0 = blockIdx.y * 32;
  int tx = threadIdx.x & 31, ty = threadIdx.x >> 5;  // 32 x 8
#pragma unroll
  for (int r = ty; r < 32; r += 8)
    tile[r][tx] = Wl[(size_t)(k0 + r) * N + n0 + tx];
  __syncthreads();
#pragma unroll
  for (int r = ty; r < 32; r += 8)
    Wtl[(size_t)(n0 + r) * K + k0 + tx] = (_Float16)tile[tx][r];
}

// ---------------------------------------------------------------------------
// Embedding gather: x[t,:] = emb[ids[t],:] * sqrt(D)  (one block per token)
// ---------------------------------------------------------------------------
__global__ __launch_bounds__(256) void embed_kernel(const int* __restrict__ ids,
                                                    const float* __restrict__ emb,
                                                    float* __restrict__ x) {
  int t = blockIdx.x;
  int tid = threadIdx.x;
  int id = ids[t];
  float4 v = ((const float4*)(emb + (size_t)id * DD))[tid];
  v.x *= 32.0f; v.y *= 32.0f; v.z *= 32.0f; v.w *= 32.0f;
  ((float4*)(x + (size_t)t * DD))[tid] = v;
}

// ---------------------------------------------------------------------------
// LayerNorm over D=1024, one block (256 thr) per token.
// Writes f32 (residual stream, optional) and f16 (GEMM input).
// ---------------------------------------------------------------------------
__global__ __launch_bounds__(256) void ln_dual(const float* __restrict__ in,
                                               const float* __restrict__ w,
                                               const float* __restrict__ b,
                                               float* __restrict__ outF,
                                               _Float16* __restrict__ outH) {
  __shared__ float red[256];
  int t = blockIdx.x, tid = threadIdx.x;
  float4 v = ((const float4*)(in + (size_t)t * DD))[tid];
  red[tid] = v.x + v.y + v.z + v.w;
  __syncthreads();
  for (int st = 128; st > 0; st >>= 1) {
    if (tid < st) red[tid] += red[tid + st];
    __syncthreads();
  }
  float mu = red[0] * (1.0f / 1024.0f);
  __syncthreads();
  float dx = v.x - mu, dy = v.y - mu, dz = v.z - mu, dw = v.w - mu;
  red[tid] = dx * dx + dy * dy + dz * dz + dw * dw;
  __syncthreads();
  for (int st = 128; st > 0; st >>= 1) {
    if (tid < st) red[tid] += red[tid + st];
    __syncthreads();
  }
  float rs = rsqrtf(red[0] * (1.0f / 1024.0f) + 1e-5f);
  int c0 = tid * 4;
  float o0 = dx * rs * w[c0]     + b[c0];
  float o1 = dy * rs * w[c0 + 1] + b[c0 + 1];
  float o2 = dz * rs * w[c0 + 2] + b[c0 + 2];
  float o3 = dw * rs * w[c0 + 3] + b[c0 + 3];
  if (outF) {
    float4 o = {o0, o1, o2, o3};
    ((float4*)(outF + (size_t)t * DD))[tid] = o;
  }
  v4h h = {(_Float16)o0, (_Float16)o1, (_Float16)o2, (_Float16)o3};
  *(v4h*)(outH + (size_t)t * DD + c0) = h;
}

// ---------------------------------------------------------------------------
// WMMA GEMM:  out[M,N] = A_h[M,K] @ Wt_h[N,K]^T (+ bias) (+ ReLU) (+ residual)
// Weights are K-major (pre-transposed) so every LDS access is a b128.
// Block = 256 threads (8 waves). Block tile 64x128, K-stage 64 (2 WMMA chunks).
// Register-stage pipelining: next K-stage is prefetched into VGPRs while the
// current stage's 8 WMMAs run, hiding global latency behind matrix math.
// OMODE: 0 = f32 row-major, 1 = f16 row-major, 2 = f16 V-transposed
//        (vt[((b*H + h)*DH + d)*S + s], for the attention V operand).
// M%64==0, N%128==0, K%64==0 (holds for all calls here).
// ---------------------------------------------------------------------------
template <int OMODE, bool RELU, bool RESID>
__global__ __launch_bounds__(256) void gemm_f16(
    const _Float16* __restrict__ A, const _Float16* __restrict__ Wt,
    const float* __restrict__ bias, const float* __restrict__ resid,
    void* __restrict__ outv, int M, int N, int K) {
  const int LD = 72;  // padded LDS stride (halfs): 64 K + 8 pad
  __shared__ _Float16 As[64 * LD];    // 64 rows (M) x 64 K
  __shared__ _Float16 Bts[128 * LD];  // 128 rows (N) x 64 K

  int tid = threadIdx.x;
  int lane = tid & 31, wv = tid >> 5;
  int half = lane >> 4, ln16 = lane & 15;
  int m0 = blockIdx.x * 64, n0 = blockIdx.y * 128;
  int mb = wv >> 2, nb = wv & 3;

  // cooperative-load indices
  int ar = tid >> 2, ak = (tid & 3) * 16;   // A: 64 rows x 64 K, 2x v8h each
  int brn = tid >> 1, bk = (tid & 1) * 32;  // B: 128 rows x 64 K, 4x v8h each

  const _Float16* agp = A + (size_t)(m0 + ar) * K + ak;
  const _Float16* bgp = Wt + (size_t)(n0 + brn) * K + bk;

  v8h pa0, pa1, pb0, pb1, pb2, pb3;
  auto gload = [&](int k0) {
    pa0 = *(const v8h*)(agp + k0);
    pa1 = *(const v8h*)(agp + k0 + 8);
    pb0 = *(const v8h*)(bgp + k0);
    pb1 = *(const v8h*)(bgp + k0 + 8);
    pb2 = *(const v8h*)(bgp + k0 + 16);
    pb3 = *(const v8h*)(bgp + k0 + 24);
  };

  v8f c00 = {}, c01 = {}, c10 = {}, c11 = {};
  gload(0);

  for (int k0 = 0; k0 < K; k0 += 64) {
    __syncthreads();
    *(v8h*)(As + ar * LD + ak)       = pa0;
    *(v8h*)(As + ar * LD + ak + 8)   = pa1;
    *(v8h*)(Bts + brn * LD + bk)      = pb0;
    *(v8h*)(Bts + brn * LD + bk + 8)  = pb1;
    *(v8h*)(Bts + brn * LD + bk + 16) = pb2;
    *(v8h*)(Bts + brn * LD + bk + 24) = pb3;
    __syncthreads();
    if (k0 + 64 < K) gload(k0 + 64);  // overlaps with WMMAs below

    const _Float16* a0p = As + (mb * 32 + ln16) * LD;
    const _Float16* a1p = a0p + 16 * LD;
    const _Float16* b0p = Bts + (nb * 32 + ln16) * LD + 16 * half;
    const _Float16* b1p = b0p + 16 * LD;
#pragma unroll
    for (int kc = 0; kc < 2; ++kc) {
      v16h a0 = ld_frag2(a0p + kc * 32 + 16 * half, a0p + kc * 32 + 8 + 16 * half);
      v16h a1 = ld_frag2(a1p + kc * 32 + 16 * half, a1p + kc * 32 + 8 + 16 * half);
      v16h b0 = ld_frag2(b0p + kc * 32, b0p + kc * 32 + 8);
      v16h b1 = ld_frag2(b1p + kc * 32, b1p + kc * 32 + 8);
      c00 = WMMA_F16(a0, b0, c00);
      c01 = WMMA_F16(a0, b1, c01);
      c10 = WMMA_F16(a1, b0, c10);
      c11 = WMMA_F16(a1, b1, c11);
    }
  }

  // Epilogue: C layout -> lane ln16 holds col, VGPR g holds row g + 8*half
  int colA = n0 + nb * 32 + ln16;
  int colB = colA + 16;
  float bvA = bias ? bias[colA] : 0.0f;
  float bvB = bias ? bias[colB] : 0.0f;
#pragma unroll
  for (int i = 0; i < 2; ++i) {
    const v8f& cl = i ? c10 : c00;
    const v8f& cr = i ? c11 : c01;
#pragma unroll
    for (int g = 0; g < 8; ++g) {
      int row = m0 + mb * 32 + i * 16 + g + 8 * half;
      float vA = cl[g] + bvA;
      float vB = cr[g] + bvB;
      if (RELU) { vA = fmaxf(vA, 0.0f); vB = fmaxf(vB, 0.0f); }
      if (RESID) {
        vA += resid[(size_t)row * N + colA];
        vB += resid[(size_t)row * N + colB];
      }
      if (OMODE == 0) {
        ((float*)outv)[(size_t)row * N + colA] = vA;
        ((float*)outv)[(size_t)row * N + colB] = vB;
      } else if (OMODE == 1) {
        ((_Float16*)outv)[(size_t)row * N + colA] = (_Float16)vA;
        ((_Float16*)outv)[(size_t)row * N + colB] = (_Float16)vB;
      } else {
        // V-transposed: vt[((b*H + h)*DH + d)*S + s]
        int bi = row >> 10, s = row & (SS - 1);
        size_t baseA = ((size_t)(bi * HH + (colA >> 6)) * DH + (colA & 63)) * SS + s;
        size_t baseB = ((size_t)(bi * HH + (colB >> 6)) * DH + (colB & 63)) * SS + s;
        ((_Float16*)outv)[baseA] = (_Float16)vA;
        ((_Float16*)outv)[baseB] = (_Float16)vB;
      }
    }
  }
}

// ---------------------------------------------------------------------------
// Flash attention with ALiBi + causal mask, all-WMMA, base-2 online softmax.
// Grid: B*H*(S/64) blocks of 128 threads (4 waves).
// Each wave owns 16 query rows; K/V streamed in 32-key LDS tiles with
// register-stage prefetch of the next tile.
// Causal structure: fully-above-diagonal chunks skip compute (barriers stay
// uniform), masking cndmasks run only on the 1-2 diagonal chunks per wave,
// and in the interior |q-k| = q-k makes ALiBi affine in kv -> one FMA/score.
// ---------------------------------------------------------------------------
__global__ __launch_bounds__(128) void attn_kernel(const _Float16* __restrict__ qh,
                                                   const _Float16* __restrict__ kh,
                                                   const _Float16* __restrict__ vt,
                                                   _Float16* __restrict__ oh) {
  const int LDK = 72, LDV = 40, LDP = 40;
  __shared__ _Float16 Kt[32 * LDK];      // [key][dim]
  __shared__ _Float16 Vts[64 * LDV];     // [dim][key]
  __shared__ _Float16 Ps[4][16 * LDP];   // per-wave P scratch

  int tid = threadIdx.x;
  int lane = tid & 31, wv = tid >> 5;
  int half = lane >> 4, ln16 = lane & 15;
  int qb = blockIdx.x & 15;          // S/64 = 16 query tiles
  int h  = (blockIdx.x >> 4) & 15;
  int b  = blockIdx.x >> 8;

  const float LOG2E  = 1.4426950408889634f;
  const float scale2 = 0.125f * LOG2E;                          // fold 1/sqrt(DH)
  const float slope2 = exp2f(-0.5f * (float)(h + 1)) * LOG2E;   // fold ALiBi slope

  // Q fragments for this wave's 16 rows (b128 loads)
  int q0 = qb * 64 + wv * 16;
  const _Float16* qrow = qh + ((size_t)(b * SS + q0 + ln16)) * DD + h * 64;
  v16h aq0 = ld_frag2(qrow + 16 * half, qrow + 8 + 16 * half);
  v16h aq1 = ld_frag2(qrow + 32 + 16 * half, qrow + 40 + 16 * half);

  v8f o0 = {}, o1 = {}, o2 = {}, o3 = {};
  float m[8], l[8], ab[8];
#pragma unroll
  for (int g = 0; g < 8; ++g) {
    m[g] = -3.0e38f;
    l[g] = 0.0f;
    // ALiBi (base-2): score = c*scale2 + ab; ab = slope2*(kpos - qpos) <= 0
    ab[g] = slope2 * (float)(ln16 - (q0 + g + 8 * half));
  }
  const float abS16 = slope2 * 16.0f;  // key band 1 offset
  const float abS32 = slope2 * 32.0f;  // per-chunk increment

  // cooperative tile-load indices
  int kk_ld = tid >> 2, cc_ld = (tid & 3) * 16;  // K: 32 keys x 64 dims
  int vd_ld = tid >> 1, vk_ld = (tid & 1) * 16;  // V^T: 64 dims x 32 keys
  const _Float16* kgp = kh + ((size_t)(b * SS + kk_ld)) * DD + h * 64 + cc_ld;
  const _Float16* vgp = vt + ((size_t)(b * HH + h) * DH + vd_ld) * SS + vk_ld;

  v8h kp0, kp1, vp0, vp1;
  auto gload_tiles = [&](int kv) {
    kp0 = *(const v8h*)(kgp + (size_t)kv * DD);
    kp1 = *(const v8h*)(kgp + (size_t)kv * DD + 8);
    vp0 = *(const v8h*)(vgp + kv);
    vp1 = *(const v8h*)(vgp + kv + 8);
  };

  int kv_end = (qb + 1) * 64;  // causal upper bound for this query tile
  gload_tiles(0);
  for (int kv = 0; kv < kv_end; kv += 32) {
    __syncthreads();
    *(v8h*)(Kt + kk_ld * LDK + cc_ld)     = kp0;
    *(v8h*)(Kt + kk_ld * LDK + cc_ld + 8) = kp1;
    *(v8h*)(Vts + vd_ld * LDV + vk_ld)     = vp0;
    *(v8h*)(Vts + vd_ld * LDV + vk_ld + 8) = vp1;
    __syncthreads();
    if (kv + 32 < kv_end) gload_tiles(kv + 32);  // overlaps with WMMAs

    // Wave-uniform causal classification for rows [q0, q0+15]:
    if (kv <= q0 + 15) {  // else: fully above diagonal -> no compute
      // S tiles = Q(16x64) @ K^T(64x32): two 16x16 C tiles, 2 WMMAs each.
      v8f c0 = {}, c1 = {};
      {
        const _Float16* k0p = Kt + ln16 * LDK + 16 * half;        // key band 0
        const _Float16* k1p = Kt + (16 + ln16) * LDK + 16 * half; // key band 1
        v16h b00 = ld_frag2(k0p, k0p + 8);            // dims 0..31
        v16h b01 = ld_frag2(k0p + 32, k0p + 40);      // dims 32..63
        v16h b10 = ld_frag2(k1p, k1p + 8);
        v16h b11 = ld_frag2(k1p + 32, k1p + 40);
        c0 = WMMA_F16(aq0, b00, c0);
        c0 = WMMA_F16(aq1, b01, c0);
        c1 = WMMA_F16(aq0, b10, c1);
        c1 = WMMA_F16(aq1, b11, c1);
      }

      bool diag = (kv + 31 > q0);  // wave-uniform: masking needed
#pragma unroll
      for (int g = 0; g < 8; ++g) {
        float s0 = fmaf(c0[g], scale2, ab[g]);
        float s1 = fmaf(c1[g], scale2, ab[g] + abS16);
        if (diag) {
          int qpos = q0 + g + 8 * half;
          if (kv + ln16 > qpos)      s0 = -3.0e38f;
          if (kv + 16 + ln16 > qpos) s1 = -3.0e38f;
        }
        float mx = fmaxf(s0, s1);
#pragma unroll
        for (int d = 8; d > 0; d >>= 1) mx = fmaxf(mx, __shfl_xor(mx, d, 16));
        float nm = fmaxf(m[g], mx);
        float f = exp2f(m[g] - nm);
        float e0 = exp2f(s0 - nm);
        float e1 = exp2f(s1 - nm);
        float rs = e0 + e1;
#pragma unroll
        for (int d = 8; d > 0; d >>= 1) rs += __shfl_xor(rs, d, 16);
        l[g] = l[g] * f + rs;
        m[g] = nm;
        o0[g] *= f; o1[g] *= f; o2[g] *= f; o3[g] *= f;
        // stash P in wave-private LDS (C layout -> row-major 16x32)
        Ps[wv][(g + 8 * half) * LDP + ln16]      = (_Float16)e0;
        Ps[wv][(g + 8 * half) * LDP + 16 + ln16] = (_Float16)e1;
      }

      // P(16x32) as A fragment (b128 loads), then O += P @ V: 4 WMMAs
      const _Float16* pr = &Ps[wv][ln16 * LDP];
      v16h pa = ld_frag2(pr + 16 * half, pr + 8 + 16 * half);
#pragma unroll
      for (int db = 0; db < 4; ++db) {
        const _Float16* vp = Vts + (db * 16 + ln16) * LDV + 16 * half;
        v16h bvf = ld_frag2(vp, vp + 8);
        v8f* op = (db == 0) ? &o0 : (db == 1) ? &o1 : (db == 2) ? &o2 : &o3;
        *op = WMMA_F16(pa, bvf, *op);
      }
    }
#pragma unroll
    for (int g = 0; g < 8; ++g) ab[g] += abS32;
  }

  // Normalize and write O
#pragma unroll
  for (int g = 0; g < 8; ++g) {
    float inv = 1.0f / l[g];
    int qpos = q0 + g + 8 * half;
    _Float16* orow = oh + ((size_t)(b * SS + qpos)) * DD + h * 64 + ln16;
    orow[0]  = (_Float16)(o0[g] * inv);
    orow[16] = (_Float16)(o1[g] * inv);
    orow[32] = (_Float16)(o2[g] * inv);
    orow[48] = (_Float16)(o3[g] * inv);
  }
}

// ---------------------------------------------------------------------------
// Host-side launch sequence
// ---------------------------------------------------------------------------
extern "C" void kernel_launch(void* const* d_in, const int* in_sizes, int n_in,
                              void* d_out, int out_size, void* d_ws, size_t ws_size,
                              hipStream_t stream) {
  const int*   ids  = (const int*)d_in[0];
  const float* emb  = (const float*)d_in[1];
  const float* Wq   = (const float*)d_in[2];
  const float* bq   = (const float*)d_in[3];
  const float* Wk   = (const float*)d_in[4];
  const float* bk   = (const float*)d_in[5];
  const float* Wv   = (const float*)d_in[6];
  const float* bv   = (const float*)d_in[7];
  const float* Wo   = (const float*)d_in[8];
  const float* bo   = (const float*)d_in[9];
  const float* W1   = (const float*)d_in[10];
  const float* b1   = (const float*)d_in[11];
  const float* W2   = (const float*)d_in[12];
  const float* b2   = (const float*)d_in[13];
  const float* ln1w = (const float*)d_in[14];
  const float* ln1b = (const float*)d_in[15];
  const float* ln2w = (const float*)d_in[16];
  const float* ln2b = (const float*)d_in[17];
  const float* lnfw = (const float*)d_in[18];
  const float* lnfb = (const float*)d_in[19];
  const float* lmw  = (const float*)d_in[20];

  // bump allocator over workspace
  char* p = (char*)d_ws;
  auto alloc = [&](size_t bytes) -> void* {
    void* r = (void*)p;
    p += (bytes + 255) & ~(size_t)255;
    return r;
  };

  const size_t DDsz = (size_t)DD * DD;   // 1M
  const size_t DFsz = (size_t)DD * FF;   // 4M

  _Float16* WqT  = (_Float16*)alloc(LL * DDsz * 2);
  _Float16* WkT  = (_Float16*)alloc(LL * DDsz * 2);
  _Float16* WvT  = (_Float16*)alloc(LL * DDsz * 2);
  _Float16* WoT  = (_Float16*)alloc(LL * DDsz * 2);
  _Float16* W1T  = (_Float16*)alloc(LL * DFsz * 2);
  _Float16* W2T  = (_Float16*)alloc(LL * DFsz * 2);
  _Float16* lmT  = (_Float16*)alloc((size_t)DD * VV * 2);
  float*    xf32 = (float*)alloc((size_t)TT * DD * 4);
  float*    resT = (float*)alloc((size_t)TT * DD * 4);
  _Float16* xh   = (_Float16*)alloc((size_t)TT * DD * 2);
  _Float16* qhb  = (_Float16*)alloc((size_t)TT * DD * 2);
  _Float16* khb  = (_Float16*)alloc((size_t)TT * DD * 2);
  _Float16* vtb  = (_Float16*)alloc((size_t)TT * DD * 2);  // V transposed [B,H,DH,S]
  _Float16* ahb  = (_Float16*)alloc((size_t)TT * DD * 2);
  _Float16* ffh  = (_Float16*)alloc((size_t)TT * FF * 2);

  // One-shot weight transpose+convert to K-major f16 (deterministic each call)
  transpose_cvt<<<dim3(DD / 32, DD / 32, LL), 256, 0, stream>>>(Wq, WqT, DD, DD);
  transpose_cvt<<<dim3(DD / 32, DD / 32, LL), 256, 0, stream>>>(Wk, WkT, DD, DD);
  transpose_cvt<<<dim3(DD / 32, DD / 32, LL), 256, 0, stream>>>(Wv, WvT, DD, DD);
  transpose_cvt<<<dim3(DD / 32, DD / 32, LL), 256, 0, stream>>>(Wo, WoT, DD, DD);
  transpose_cvt<<<dim3(DD / 32, FF / 32, LL), 256, 0, stream>>>(W1, W1T, DD, FF);
  transpose_cvt<<<dim3(FF / 32, DD / 32, LL), 256, 0, stream>>>(W2, W2T, FF, DD);
  transpose_cvt<<<dim3(DD / 32, VV / 32, 1), 256, 0, stream>>>(lmw, lmT, DD, VV);

  // Embedding
  embed_kernel<<<TT, 256, 0, stream>>>(ids, emb, xf32);
  cvt_f32_f16<<<(TT * DD / 4 + 255) / 256, 256, 0, stream>>>(xf32, xh, TT * DD);

  dim3 gD(TT / 64, DD / 128);
  dim3 gF(TT / 64, FF / 128);

  for (int l = 0; l < LL; ++l) {
    // QKV projections (f16 out; V written pre-transposed for attention)
    gemm_f16<1, false, false><<<gD, 256, 0, stream>>>(
        xh, WqT + (size_t)l * DDsz, bq + l * DD, nullptr, qhb, TT, DD, DD);
    gemm_f16<1, false, false><<<gD, 256, 0, stream>>>(
        xh, WkT + (size_t)l * DDsz, bk + l * DD, nullptr, khb, TT, DD, DD);
    gemm_f16<2, false, false><<<gD, 256, 0, stream>>>(
        xh, WvT + (size_t)l * DDsz, bv + l * DD, nullptr, vtb, TT, DD, DD);

    // Flash attention (ALiBi + causal)
    attn_kernel<<<BB * HH * (SS / 64), 128, 0, stream>>>(qhb, khb, vtb, ahb);

    // O projection + residual -> resT (f32)
    gemm_f16<0, false, true><<<gD, 256, 0, stream>>>(
        ahb, WoT + (size_t)l * DDsz, bo + l * DD, xf32, resT, TT, DD, DD);
    // ln1 -> xf32 (residual stream) + xh (GEMM input)
    ln_dual<<<TT, 256, 0, stream>>>(resT, ln1w + l * DD, ln1b + l * DD, xf32, xh);

    // FFN1 (ReLU, f16 out), FFN2 (+residual, f32 out)
    gemm_f16<1, true, false><<<gF, 256, 0, stream>>>(
        xh, W1T + (size_t)l * DFsz, b1 + l * FF, nullptr, ffh, TT, FF, DD);
    gemm_f16<0, false, true><<<gD, 256, 0, stream>>>(
        ffh, W2T + (size_t)l * DFsz, b2 + l * DD, xf32, resT, TT, DD, FF);
    // ln2 -> xf32 + xh
    ln_dual<<<TT, 256, 0, stream>>>(resT, ln2w + l * DD, ln2b + l * DD, xf32, xh);
  }

  // Final LN (f16 only) + LM head -> d_out (f32 logits)
  ln_dual<<<TT, 256, 0, stream>>>(xf32, lnfw, lnfb, nullptr, xh);
  dim3 gV(TT / 64, VV / 128);
  gemm_f16<0, false, false><<<gV, 256, 0, stream>>>(
      xh, lmT, nullptr, nullptr, d_out, TT, VV, DD);
}
